// MiniAttn_7868380086520
// MI455X (gfx1250) — compile-verified, matrix operand order checked by reading, not run on hardware
//
#include <hip/hip_runtime.h>
#include <cstdint>
#include <math.h>

// Problem constants (from the reference): B=2, N=8192, D=768
#define BB 2
#define NN 8192
#define DD 768
#define NPAIR 384           // D/2 rotary pairs
#define PER 128             // pairs per angle channel (384/3)
#define NW 8                // wave32s per 256-thread block
#define WCOLS 96            // D columns owned by each wave (8*96 = 768)
#define NTILE 6             // 16-wide WMMA N-tiles per wave (6*16 = 96)
#define CHUNK 32            // keys per flash-attention chunk (= WMMA K for f16)
#define QROW 776            // padded LDS row stride (halves): 388 dwords, bank-friendly
#define XROW 776
#define VROW 776
#define PROW 40             // padded P row stride (halves)

typedef __attribute__((ext_vector_type(16))) _Float16 v16h;
typedef __attribute__((ext_vector_type(8)))  float    v8f;
typedef unsigned int uintx4 __attribute__((ext_vector_type(4)));
typedef int          intx8  __attribute__((ext_vector_type(8)));
typedef int          intx4  __attribute__((ext_vector_type(4)));

union Frag16 { v16h v; uint32_t u[8]; _Float16 h[16]; };

// CDNA5 16-bit A/B fragment K-pattern (ISA 7.12.2): reg r holds K-pair
//   r<4 : K = kb + 2r      (kb = 0 for lanes 0-15, 8 for lanes 16-31)
//   r>=4: K = 16 + kb + 2(r-4)
__device__ __forceinline__ int kpair_base(int r, int kb) {
  return (r < 4) ? (kb + 2 * r) : (16 + kb + 2 * (r - 4));
}

__device__ __forceinline__ v8f wmma_f16(const Frag16& a, const Frag16& b, v8f c) {
  return __builtin_amdgcn_wmma_f32_16x16x32_f16(
      /*neg_a=*/false, a.v, /*neg_b=*/false, b.v,
      /*c_mod=*/(short)0, c, /*reuse_a=*/false, /*reuse_b=*/false);
}

// Generic->LDS byte offset (low 32 bits of the shared aperture address).
__device__ __forceinline__ uint32_t lds_off(const void* p) {
  return (uint32_t)(uintptr_t)p;
}

// Per-lane async global->LDS 16B copy (ASYNCcnt path, no VGPR round trip).
__device__ __forceinline__ void async_b128(uint32_t lds_byte_off, uint64_t gaddr) {
  asm volatile("global_load_async_to_lds_b128 %0, %1, off"
               :: "v"(lds_byte_off), "v"(gaddr) : "memory");
}
__device__ __forceinline__ void wait_async0() {
  asm volatile("s_wait_asynccnt 0x0" ::: "memory");
}

// TDM: DMA one [rows x 768] fp16 tile (row-major, contiguous) into LDS.
// D# per ISA ch.8: group0 = {count, lds_addr, global_addr, type=2},
// group1 = {data_size=2B, dims/strides}, groups 2/3 unused (2-D tile).
// This toolchain's builtin takes 6 args (extra int32x8 before cpol).
__device__ __forceinline__ void tdm_load_rows_f16(uint32_t ldsaddr, const void* gsrc,
                                                  int rows) {
  uint64_t ga = (uint64_t)(uintptr_t)gsrc;
  uintx4 g0;
  g0.x = 1u;                                   // count=1 (valid descriptor)
  g0.y = ldsaddr;                              // LDS byte address
  g0.z = (uint32_t)ga;                         // global_addr[31:0]
  g0.w = (uint32_t)((ga >> 32) & 0x01FFFFFFu) | (2u << 30);  // addr[56:32] | type=2
  intx8 g1;
  g1[0] = 0x00010000;                          // data_size=1 -> 2-byte elements
  g1[1] = (int)(768u << 16);                   // tensor_dim0 = 768 (low 16)
  g1[2] = (int)(8192u << 16);                  // tensor_dim1 = 8192 (low 16)
  g1[3] = (int)(768u << 16);                   // tile_dim0 = 768
  g1[4] = rows;                                // tile_dim1 = rows, tile_dim2 = 0
  g1[5] = 768;                                 // tensor_dim0_stride = 768
  g1[6] = 0;                                   // dim1_stride low16<<16 (0x600000 -> 0)
  g1[7] = 96;                                  // dim1_stride[47:16] = 768*8192>>16
  __builtin_amdgcn_tensor_load_to_lds(g0, g1, (intx4)0, (intx4)0, (intx8)0, 0);
}

// ---------------------------------------------------------------------------
// Kernel 0: fp32 -> fp16 conversion of x and the four weight matrices.
// ---------------------------------------------------------------------------
__global__ __launch_bounds__(256) void convert_kernel(
    const float* __restrict__ x,
    const float* __restrict__ Wq, const float* __restrict__ Wk,
    const float* __restrict__ Wv, const float* __restrict__ Wo,
    _Float16* __restrict__ xh, _Float16* __restrict__ wh) {
  const size_t nx = (size_t)BB * NN * DD;
  const size_t nw = (size_t)DD * DD;
  const size_t total = nx + 4 * nw;
  size_t stride = (size_t)gridDim.x * 256u;
  for (size_t i = (size_t)blockIdx.x * 256u + threadIdx.x; i < total; i += stride) {
    if (i < nx) {
      xh[i] = (_Float16)x[i];
    } else {
      size_t wi = i - nx;
      int m = (int)(wi / nw);
      size_t o = wi % nw;
      const float* src = (m == 0) ? Wq : (m == 1) ? Wk : (m == 2) ? Wv : Wo;
      wh[(size_t)m * nw + o] = (_Float16)src[o];
    }
  }
}

// ---------------------------------------------------------------------------
// Kernel 1: row-tile GEMM  out[16 tokens, 768] = A[16,768] * W^T + bias
//   mode 0: fused S2-RoPE epilogue, fp16 output (q / k)
//   mode 1: bias only, fp16 output (v)
//   mode 2: bias only, fp32 output (final projection -> d_out)
// ---------------------------------------------------------------------------
__global__ __launch_bounds__(256) void qkv_gemm_kernel(
    const _Float16* __restrict__ A, const _Float16* __restrict__ W,
    const float* __restrict__ bias, const float* __restrict__ pos,
    const float* __restrict__ vd, const float* __restrict__ ref,
    void* __restrict__ dst, int mode) {
  __shared__ alignas(16) _Float16 xs[16 * XROW];   // ~24 KB, padded rows
  __shared__ alignas(16) float    outs[16 * DD];   // 48 KB fp32 result tile
  __shared__ float ang[3 * 16];

  const int t = threadIdx.x;
  const int lane = t & 31;
  const int w = t >> 5;
  const int b = blockIdx.x / (NN / 16);
  const int mt = blockIdx.x % (NN / 16);
  const size_t row0 = (size_t)b * NN + (size_t)mt * 16;

  {  // A tile -> padded LDS rows
    const uint4* src = (const uint4*)(A + row0 * DD);
#pragma unroll
    for (int i = 0; i < 6; ++i) {
      int hbase = (t + i * 256) * 8;
      int row = hbase / DD, col = hbase % DD;
      *(uint4*)&xs[row * XROW + col] = src[t + i * 256];
    }
  }
  if (mode == 0 && t < 16) {
    size_t tok = row0 + t;
    float th = pos[tok * 2 + 0], ph = pos[tok * 2 + 1];
    float vx = vd[tok * 3 + 0], vy = vd[tok * 3 + 1], vz = vd[tok * 3 + 2];
    float rx = ref[b * 3 + 0], ry = ref[b * 3 + 1], rz = ref[b * 3 + 2];
    float vn = rsqrtf(vx * vx + vy * vy + vz * vz);
    float rn = rsqrtf(rx * rx + ry * ry + rz * rz);
    float dp = (vx * rx + vy * ry + vz * rz) * vn * rn;
    dp = fminf(1.f, fmaxf(-1.f, dp));
    ang[t] = th; ang[16 + t] = ph; ang[32 + t] = acosf(dp);
  }
  __syncthreads();

  const int hi = lane >> 4;
  const int kb = hi * 8;
  const int ln = lane & 15;

  const v8f vzero = {0.f, 0.f, 0.f, 0.f, 0.f, 0.f, 0.f, 0.f};
  v8f acc[NTILE];
#pragma unroll
  for (int j = 0; j < NTILE; ++j) acc[j] = vzero;

  for (int kc = 0; kc < DD; kc += 32) {
    Frag16 a;
#pragma unroll
    for (int r = 0; r < 8; ++r)
      a.u[r] = *(const uint32_t*)&xs[ln * XROW + kc + kpair_base(r, kb)];
#pragma unroll
    for (int j = 0; j < NTILE; ++j) {
      const int jcol = w * WCOLS + j * 16 + ln;
      Frag16 bf;
#pragma unroll
      for (int r = 0; r < 8; ++r)
        bf.u[r] = *(const uint32_t*)&W[(size_t)jcol * DD + kc + kpair_base(r, kb)];
      acc[j] = wmma_f16(a, bf, acc[j]);
    }
  }

#pragma unroll
  for (int j = 0; j < NTILE; ++j) {
    const int jcol = w * WCOLS + j * 16 + ln;
#pragma unroll
    for (int r = 0; r < 8; ++r)
      outs[(r + 8 * hi) * DD + jcol] = acc[j][r];
  }
  __syncthreads();

  const int rr = t >> 4;
  const int cc = t & 15;
  if (mode == 0) {
    _Float16* out = (_Float16*)dst;
    const float l2b_per = 13.2877124f / (float)PER;  // log2(10000)/128
    float th = ang[rr], ph = ang[16 + rr], ps = ang[32 + rr];
    size_t o = (row0 + rr) * DD;
    for (int i = cc; i < NPAIR; i += 16) {
      int ch = i / PER, f = i % PER;
      float base = (ch == 0) ? th : (ch == 1) ? ph : ps;
      float angf = base * exp2f(-(float)f * l2b_per);
      float s = __sinf(angf), c = __cosf(angf);
      float t1 = outs[rr * DD + i] + bias[i];
      float t2 = outs[rr * DD + i + NPAIR] + bias[i + NPAIR];
      out[o + i]         = (_Float16)(t1 * c - t2 * s);
      out[o + i + NPAIR] = (_Float16)(t1 * s + t2 * c);
    }
  } else if (mode == 1) {
    _Float16* out = (_Float16*)dst;
    size_t o = (row0 + rr) * DD;
    for (int i = cc; i < DD; i += 16)
      out[o + i] = (_Float16)(outs[rr * DD + i] + bias[i]);
  } else {
    float* out = (float*)dst;
    size_t o = (row0 + rr) * DD;
    for (int i = cc; i < DD; i += 16)
      out[o + i] = outs[rr * DD + i] + bias[i];
  }
}

// ---------------------------------------------------------------------------
// Kernel 2: flash attention. K chunk via TDM DMA (TENSORcnt), V chunk via
// per-lane async global->LDS b128 (ASYNCcnt), fp32 online softmax, all GEMMs
// on v_wmma_f32_16x16x32_f16.
// ---------------------------------------------------------------------------
__global__ __launch_bounds__(256) void attn_kernel(
    const _Float16* __restrict__ Q, const _Float16* __restrict__ K,
    const _Float16* __restrict__ V, _Float16* __restrict__ O) {
  __shared__ alignas(16) _Float16 qs[16 * QROW];        // ~24 KB, padded
  __shared__ alignas(16) _Float16 ks[CHUNK * DD];       // 48 KB, TDM target
  __shared__ alignas(16) _Float16 vs[CHUNK * VROW];     // ~49 KB, padded, async
  __shared__ alignas(16) float Spart[NW * 16 * CHUNK];  // 16 KB per-wave partials
  __shared__ float Sf[16 * CHUNK];
  __shared__ _Float16 Pf[16 * PROW];
  __shared__ float mrow[16], lrow[16], arow[16];

  const int t = threadIdx.x;
  const int lane = t & 31;
  const int w = t >> 5;
  const int b = blockIdx.x / (NN / 16);
  const int mt = blockIdx.x % (NN / 16);
  const size_t q0 = ((size_t)b * NN + (size_t)mt * 16) * DD;
  const int hi = lane >> 4;
  const int kb = hi * 8;
  const int ln = lane & 15;
  const float scale = 0.0360843918f;  // 1/sqrt(768)

  {  // Q tile -> padded LDS rows
    const uint4* src = (const uint4*)(Q + q0);
#pragma unroll
    for (int i = 0; i < 6; ++i) {
      int hbase = (t + i * 256) * 8;
      int row = hbase / DD, col = hbase % DD;
      *(uint4*)&qs[row * QROW + col] = src[t + i * 256];
    }
  }
  if (t < 16) { mrow[t] = -1e30f; lrow[t] = 0.f; }

  const v8f vzero = {0.f, 0.f, 0.f, 0.f, 0.f, 0.f, 0.f, 0.f};
  v8f o[NTILE];
#pragma unroll
  for (int j = 0; j < NTILE; ++j) o[j] = vzero;

  const uint32_t ksOff = lds_off(&ks[0]);
  const uint32_t vsOff = lds_off(&vs[0]);
  const size_t kvbase = (size_t)b * NN * DD;

  for (int c = 0; c < NN / CHUNK; ++c) {
    __syncthreads();  // previous iteration done reading ks/vs/Pf
    const size_t coff = kvbase + (size_t)c * CHUNK * DD;

    if (w == 0)  // one TDM descriptor pulls the whole 32x768 fp16 K tile
      tdm_load_rows_f16(ksOff, K + coff, CHUNK);

    {  // V chunk: per-lane async copies into padded LDS rows
      const uint64_t vbase = (uint64_t)(uintptr_t)(V + coff);
#pragma unroll
      for (int i = 0; i < 12; ++i) {
        int hbase = (t + i * 256) * 8;
        int key = hbase / DD, d0 = hbase % DD;
        async_b128(vsOff + (uint32_t)(key * VROW + d0) * 2u,
                   vbase + (uint64_t)hbase * 2u);
      }
    }
    if (c + 1 < NN / CHUNK) {  // warm GL2 for next chunk
      __builtin_prefetch(K + coff + (size_t)CHUNK * DD + (size_t)t * 96, 0, 1);
      __builtin_prefetch(V + coff + (size_t)CHUNK * DD + (size_t)t * 96, 0, 1);
    }
    wait_async0();                               // own wave's async V copies
    if (w == 0) __builtin_amdgcn_s_wait_tensorcnt(0);  // TDM K tile done
    __syncthreads();

    // Partial S = Q[:, wslice] * K[:, wslice]^T  (two 16x16 accumulators)
    v8f s0 = vzero, s1 = vzero;
#pragma unroll
    for (int kk = 0; kk < 3; ++kk) {
      const int d0 = w * WCOLS + kk * 32;
      Frag16 a, b0, b1;
#pragma unroll
      for (int r = 0; r < 8; ++r) {
        int kp = kpair_base(r, kb);
        a.u[r]  = *(const uint32_t*)&qs[ln * QROW + d0 + kp];
        b0.u[r] = *(const uint32_t*)&ks[ln * DD + d0 + kp];
        b1.u[r] = *(const uint32_t*)&ks[(ln + 16) * DD + d0 + kp];
      }
      s0 = wmma_f16(a, b0, s0);
      s1 = wmma_f16(a, b1, s1);
    }
#pragma unroll
    for (int r = 0; r < 8; ++r) {
      int m = r + 8 * hi;
      Spart[(w * 16 + m) * CHUNK + ln]      = s0[r];
      Spart[(w * 16 + m) * CHUNK + 16 + ln] = s1[r];
    }
    __syncthreads();
#pragma unroll
    for (int e = t; e < 16 * CHUNK; e += 256) {
      float s = 0.f;
#pragma unroll
      for (int ww = 0; ww < NW; ++ww) s += Spart[ww * 16 * CHUNK + e];
      Sf[e] = s;
    }
    __syncthreads();

    // Online softmax (fp32), one thread per query row.
    if (t < 16) {
      float mold = mrow[t];
      float mx = mold;
      for (int jx = 0; jx < CHUNK; ++jx) mx = fmaxf(mx, Sf[t * CHUNK + jx] * scale);
      float al = __expf(mold - mx);
      float ssum = 0.f;
      for (int jx = 0; jx < CHUNK; ++jx) {
        float p = __expf(Sf[t * CHUNK + jx] * scale - mx);
        Pf[t * PROW + jx] = (_Float16)p;
        ssum += p;
      }
      mrow[t] = mx; arow[t] = al; lrow[t] = lrow[t] * al + ssum;
    }
    __syncthreads();

    // O = O*alpha + P*V   (V^T fragments gathered from row-major vs)
    Frag16 pa;
#pragma unroll
    for (int r = 0; r < 8; ++r)
      pa.u[r] = *(const uint32_t*)&Pf[ln * PROW + kpair_base(r, kb)];
#pragma unroll
    for (int j = 0; j < NTILE; ++j) {
      const int dcol = w * WCOLS + j * 16 + ln;
      Frag16 vb;
#pragma unroll
      for (int r = 0; r < 8; ++r) {
        int kp = kpair_base(r, kb);
        vb.h[2 * r]     = vs[kp * VROW + dcol];
        vb.h[2 * r + 1] = vs[(kp + 1) * VROW + dcol];
      }
#pragma unroll
      for (int r = 0; r < 8; ++r) o[j][r] *= arow[r + 8 * hi];
      o[j] = wmma_f16(pa, vb, o[j]);
    }
  }
  __syncthreads();

#pragma unroll
  for (int j = 0; j < NTILE; ++j) {
    const int dcol = w * WCOLS + j * 16 + ln;
#pragma unroll
    for (int r = 0; r < 8; ++r) {
      int m = r + 8 * hi;
      O[q0 + (size_t)m * DD + dcol] = (_Float16)(o[j][r] / lrow[m]);
    }
  }
}

// ---------------------------------------------------------------------------
// Host-side launch. Workspace (fp16 halves):
//   xh[B*N*D] | wh[4*D*D] | qh | kh | vh | oh  => ~131 MB.
// ---------------------------------------------------------------------------
extern "C" void kernel_launch(void* const* d_in, const int* in_sizes, int n_in,
                              void* d_out, int out_size, void* d_ws, size_t ws_size,
                              hipStream_t stream) {
  const float* x   = (const float*)d_in[0];
  const float* pos = (const float*)d_in[1];
  const float* vd  = (const float*)d_in[2];
  const float* ref = (const float*)d_in[3];
  const float* Wq  = (const float*)d_in[4];
  const float* bq  = (const float*)d_in[5];
  const float* Wk  = (const float*)d_in[6];
  const float* bk  = (const float*)d_in[7];
  const float* Wv  = (const float*)d_in[8];
  const float* bv  = (const float*)d_in[9];
  const float* Wo  = (const float*)d_in[10];
  const float* bo  = (const float*)d_in[11];

  const size_t BND = (size_t)BB * NN * DD;
  const size_t WSZ = (size_t)DD * DD;
  _Float16* xh = (_Float16*)d_ws;
  _Float16* wh = xh + BND;
  _Float16* qh = wh + 4 * WSZ;
  _Float16* kh = qh + BND;
  _Float16* vh = kh + BND;
  _Float16* oh = vh + BND;

  convert_kernel<<<4096, 256, 0, stream>>>(x, Wq, Wk, Wv, Wo, xh, wh);

  const int gblocks = BB * (NN / 16);  // 1024
  qkv_gemm_kernel<<<gblocks, 256, 0, stream>>>(xh, wh,           bq, pos, vd, ref, (void*)qh, 0);
  qkv_gemm_kernel<<<gblocks, 256, 0, stream>>>(xh, wh + WSZ,     bk, pos, vd, ref, (void*)kh, 0);
  qkv_gemm_kernel<<<gblocks, 256, 0, stream>>>(xh, wh + 2 * WSZ, bv, pos, vd, ref, (void*)vh, 1);
  attn_kernel<<<gblocks, 256, 0, stream>>>(qh, kh, vh, oh);
  qkv_gemm_kernel<<<gblocks, 256, 0, stream>>>(oh, wh + 3 * WSZ, bo, pos, vd, ref, d_out, 2);
}
